// Attention_58033598104213
// MI455X (gfx1250) — compile-verified, matrix operand order checked by reading, not run on hardware
//
#include <hip/hip_runtime.h>

// ---------------------------------------------------------------------------
// MI455X (gfx1250) GQA attention: bf16 WMMA everywhere.
//   B=2, T=2048, H=16, HKV=4, HD=128, d_model=2048
// Pipeline:
//   0) convert x -> bf16;  convert+transpose wq/wk/wv/wo -> bf16 [N,K]
//   1) gemm_wmma_bf16 (async-to-LDS double-buffered):
//        Q = x@wq (bf16), K = x@wk (bf16), V^T = (x@wv)^T (bf16)
//   2) rope_kernel on Q and K (bf16 in-place)
//   3) flash_attn_wmma: online-softmax causal attention, WMMA for QK^T and PV
//   4) gemm_wmma_bf16: out = attn @ wo (f32 out)
// ---------------------------------------------------------------------------

typedef __bf16 v16bf __attribute__((ext_vector_type(16)));
typedef float  v8f   __attribute__((ext_vector_type(8)));

#define DEVINL __device__ __forceinline__

DEVINL unsigned short f32_to_bf16_raw(float f) {
  union { float f; unsigned u; } c; c.f = f;
  unsigned r = c.u + 0x7FFFu + ((c.u >> 16) & 1u);   // round-to-nearest-even
  return (unsigned short)(r >> 16);
}
DEVINL unsigned pack_bf16x2(float lo, float hi) {
  return (unsigned)f32_to_bf16_raw(lo) | ((unsigned)f32_to_bf16_raw(hi) << 16);
}
DEVINL float bf16_raw_to_f32(unsigned short s) {
  union { unsigned u; float f; } c; c.u = ((unsigned)s) << 16;
  return c.f;
}

union AFrag { v16bf v; unsigned u[8]; uint4 q[2]; };   // 16 bf16 = 8 VGPRs
union CFrag { v8f  v; float   f[8]; };                 // 16x16 f32 accum tile

// ---------------------------------------------------------------------------
// One-time conversions (bandwidth-trivial; removes per-tile cvt from GEMMs)
// ---------------------------------------------------------------------------
__global__ __launch_bounds__(256)
void convert_bf16(const float* __restrict__ src, unsigned* __restrict__ dst,
                  int total_pairs)
{
  int i = blockIdx.x * 256 + threadIdx.x;
  if (i < total_pairs) dst[i] = pack_bf16x2(src[2 * i], src[2 * i + 1]);
}

// src: f32 [K,N] row-major  ->  dst: bf16 [N,K] row-major (32x32 LDS tiles)
__global__ __launch_bounds__(256)
void convert_transpose_bf16(const float* __restrict__ src,
                            unsigned short* __restrict__ dst, int K, int N)
{
  __shared__ float tile[32][33];
  const int k0 = blockIdx.y * 32, n0 = blockIdx.x * 32;
  const int c = threadIdx.x & 31, r = threadIdx.x >> 5;   // 8-row strips
#pragma unroll
  for (int rr = 0; rr < 32; rr += 8)
    tile[rr + r][c] = src[(size_t)(k0 + rr + r) * N + n0 + c];
  __syncthreads();
#pragma unroll
  for (int rr = 0; rr < 32; rr += 8)
    dst[(size_t)(n0 + rr + r) * K + k0 + c] = f32_to_bf16_raw(tile[c][rr + r]);
}

// ---------------------------------------------------------------------------
// GEMM: C[M,N] = A[M,K] * Bt[N,K]^T,  A/Bt bf16 row-major.
// Block 256 thr (8 waves), tile 64x64, BK=32; wave w: col (w&3)*16,
// rows (w>>2)*32..+32 (two stacked 16x16 accumulators sharing the B-frag).
// Tiles staged with GLOBAL_LOAD_ASYNC_TO_LDS_B128 (ASYNCcnt) and
// double-buffered: prefetch tile t+1 while computing tile t.
// LDS rows padded to 20 uints (80B) -> aligned b128 fragment reads.
// ---------------------------------------------------------------------------
template<int OUT_BF16, int OUT_TRANS>
__global__ __launch_bounds__(256)
void gemm_wmma_bf16(const unsigned short* __restrict__ A,    // [M,K] bf16
                    const unsigned short* __restrict__ Bt,   // [N,K] bf16
                    void* __restrict__ Cp, int M, int N, int K)
{
  __shared__ __align__(16) unsigned As_u[2][64 * 20];
  __shared__ __align__(16) unsigned Bs_u[2][64 * 20];

  const int tileM = blockIdx.y * 64;
  const int tileN = blockIdx.x * 64;
  const int tid  = threadIdx.x;
  const int lane = tid & 31;
  const int w    = tid >> 5;
  const int m    = lane & 15;
  const int hi   = lane >> 4;
  const int ncol    = (w & 3) * 16;
  const int rowbase = (w >> 2) * 32;

  // per-thread async-copy assignment: one 16B chunk of A and of Bt per tile
  const int crow = tid >> 2;   // 0..63 (tile row / tile col)
  const int cchk = tid & 3;    // 16B chunk within the 64B (32 bf16) row

  const unsigned long long gA = (unsigned long long)(uintptr_t)
      (A  + (size_t)(tileM + crow) * (size_t)K + cchk * 8);
  const unsigned long long gB = (unsigned long long)(uintptr_t)
      (Bt + (size_t)(tileN + crow) * (size_t)K + cchk * 8);

  CFrag c0, c1;
#pragma unroll
  for (int r = 0; r < 8; ++r) { c0.f[r] = 0.f; c1.f[r] = 0.f; }

  auto issue_tile = [&](int buf, int k0) {
    unsigned la = (unsigned)(uintptr_t)&As_u[buf][crow * 20 + cchk * 4];
    unsigned lb = (unsigned)(uintptr_t)&Bs_u[buf][crow * 20 + cchk * 4];
    unsigned long long ga = gA + (unsigned long long)k0 * 2ull;
    unsigned long long gb = gB + (unsigned long long)k0 * 2ull;
    asm volatile("global_load_async_to_lds_b128 %0, %1, off"
                 :: "v"(la), "v"(ga) : "memory");
    asm volatile("global_load_async_to_lds_b128 %0, %1, off"
                 :: "v"(lb), "v"(gb) : "memory");
  };

  const int nk = K >> 5;
  issue_tile(0, 0);

  for (int it = 0; it < nk; ++it) {
    const int buf = it & 1;
    if (it + 1 < nk) {
      issue_tile(buf ^ 1, (it + 1) << 5);
      asm volatile("s_wait_asynccnt 2" ::: "memory");  // in-order: tile t landed
    } else {
      asm volatile("s_wait_asynccnt 0" ::: "memory");
    }
    __syncthreads();

    AFrag a0, a1, bb;
    a0.q[0] = *(const uint4*)&As_u[buf][(rowbase +      m) * 20 +     hi * 4];
    a0.q[1] = *(const uint4*)&As_u[buf][(rowbase +      m) * 20 + 8 + hi * 4];
    a1.q[0] = *(const uint4*)&As_u[buf][(rowbase + 16 + m) * 20 +     hi * 4];
    a1.q[1] = *(const uint4*)&As_u[buf][(rowbase + 16 + m) * 20 + 8 + hi * 4];
    bb.q[0] = *(const uint4*)&Bs_u[buf][(ncol + m) * 20 +     hi * 4];
    bb.q[1] = *(const uint4*)&Bs_u[buf][(ncol + m) * 20 + 8 + hi * 4];

    c0.v = __builtin_amdgcn_wmma_f32_16x16x32_bf16(false, a0.v, false, bb.v,
                                                   (short)0, c0.v, false, false);
    c1.v = __builtin_amdgcn_wmma_f32_16x16x32_bf16(false, a1.v, false, bb.v,
                                                   (short)0, c1.v, false, false);
    __syncthreads();   // reads done before next async writes into buf
  }

  // store C (C-layout: lane n = L&15, VGPR r -> row hi*8 + r)
  const int gn = tileN + ncol + m;
#pragma unroll
  for (int r = 0; r < 8; ++r) {
    int gm0 = tileM + rowbase + hi * 8 + r;
    int gm1 = gm0 + 16;
    if (OUT_BF16) {
      unsigned short* Cs = (unsigned short*)Cp;
      if (OUT_TRANS) {
        Cs[(size_t)gn * (size_t)M + gm0] = f32_to_bf16_raw(c0.f[r]);
        Cs[(size_t)gn * (size_t)M + gm1] = f32_to_bf16_raw(c1.f[r]);
      } else {
        Cs[(size_t)gm0 * (size_t)N + gn] = f32_to_bf16_raw(c0.f[r]);
        Cs[(size_t)gm1 * (size_t)N + gn] = f32_to_bf16_raw(c1.f[r]);
      }
    } else {
      float* Cf = (float*)Cp;
      Cf[(size_t)gm0 * (size_t)N + gn] = c0.f[r];
      Cf[(size_t)gm1 * (size_t)N + gn] = c1.f[r];
    }
  }
}

// ---------------------------------------------------------------------------
// RoPE (GPT-NeoX half-rotation) in-place on bf16 buffer [B,T,H,128].
// ---------------------------------------------------------------------------
__global__ __launch_bounds__(256)
void rope_kernel(unsigned short* __restrict__ buf, int H, int total)
{
  int idx = blockIdx.x * 256 + threadIdx.x;
  if (idx >= total) return;
  int d    = idx & 63;
  int rest = idx >> 6;
  int h    = rest % H;
  int pos  = rest / H;                 // b*T + t
  int t    = pos & 2047;               // T = 2048
  size_t base = ((size_t)pos * H + h) * 128;

  float ang = (float)t * __powf(10000.0f, -(float)(2 * d) * (1.0f / 128.0f));
  float s, c;
  __sincosf(ang, &s, &c);
  float x1 = bf16_raw_to_f32(buf[base + d]);
  float x2 = bf16_raw_to_f32(buf[base + 64 + d]);
  buf[base + d]      = f32_to_bf16_raw(x1 * c - x2 * s);
  buf[base + 64 + d] = f32_to_bf16_raw(x2 * c + x1 * s);
}

// ---------------------------------------------------------------------------
// Flash attention, 1 wave per block, one 16-row query tile. Resident Q
// fragments; per 32-key block: 8 WMMA (S=QK^T), online softmax (16-lane
// shuffle reductions), P C->A transpose through LDS, 8 WMMA (O+=PV).
// V pre-transposed so PV B-fragments are contiguous b128 loads.
// ---------------------------------------------------------------------------
__global__ __launch_bounds__(32)
void flash_attn_wmma(const unsigned* __restrict__ q_u,
                     const unsigned* __restrict__ k_u,
                     const unsigned* __restrict__ vT_u,
                     unsigned short* __restrict__ out_bf16)
{
  constexpr int T = 2048, H = 16, HKV = 4, HD = 128;
  constexpr float SCALE = 0.08838834764831845f;   // 1/sqrt(128)

  __shared__ __align__(16) unsigned Plds_u[16 * 20];          // 16x32 bf16, 80B rows
  unsigned short* Plds_s = (unsigned short*)Plds_u;           // row stride 40 shorts

  const int blk  = blockIdx.x;       // ((b*H + h)*128 + qt)
  const int qt   = blk & 127;
  const int h    = (blk >> 7) & 15;
  const int b    = blk >> 11;
  const int hkv  = h >> 2;
  const int qt16 = qt * 16;

  const int lane = threadIdx.x;
  const int m  = lane & 15;
  const int hi = lane >> 4;

  const unsigned* qrow = q_u + ((size_t)(b * T + qt16 + m) * H + h) * 64;
  AFrag Qf[4];
#pragma unroll
  for (int f = 0; f < 4; ++f) {
    Qf[f].q[0] = *(const uint4*)(qrow + f * 16 +     hi * 4);
    Qf[f].q[1] = *(const uint4*)(qrow + f * 16 + 8 + hi * 4);
  }

  CFrag O[8];
  float m_i[8], l_i[8];
#pragma unroll
  for (int r = 0; r < 8; ++r) { m_i[r] = -1e30f; l_i[r] = 0.f; }
#pragma unroll
  for (int dt = 0; dt < 8; ++dt)
#pragma unroll
    for (int r = 0; r < 8; ++r) O[dt].f[r] = 0.f;

  const int nblocks = (qt16 + 16 + 31) >> 5;   // causal: keys [0, qt16+16)
  for (int jb = 0; jb < nblocks; ++jb) {
    const int j0 = jb * 32;

    CFrag S0, S1;
#pragma unroll
    for (int r = 0; r < 8; ++r) { S0.f[r] = 0.f; S1.f[r] = 0.f; }

    const unsigned* kr0 = k_u + ((size_t)(b * T + j0 + m) * HKV + hkv) * 64;
    const unsigned* kr1 = kr0 + (size_t)16 * HKV * 64;
    if (jb + 1 < nblocks) {            // prefetch next key block (L2 warm)
      __builtin_prefetch((const void*)(kr0 + (size_t)32 * HKV * 64), 0, 1);
      __builtin_prefetch((const void*)(kr1 + (size_t)32 * HKV * 64), 0, 1);
    }
#pragma unroll
    for (int f = 0; f < 4; ++f) {
      AFrag kf0, kf1;
      kf0.q[0] = *(const uint4*)(kr0 + f * 16 +     hi * 4);
      kf0.q[1] = *(const uint4*)(kr0 + f * 16 + 8 + hi * 4);
      kf1.q[0] = *(const uint4*)(kr1 + f * 16 +     hi * 4);
      kf1.q[1] = *(const uint4*)(kr1 + f * 16 + 8 + hi * 4);
      S0.v = __builtin_amdgcn_wmma_f32_16x16x32_bf16(false, Qf[f].v, false, kf0.v,
                                                     (short)0, S0.v, false, false);
      S1.v = __builtin_amdgcn_wmma_f32_16x16x32_bf16(false, Qf[f].v, false, kf1.v,
                                                     (short)0, S1.v, false, false);
    }

#pragma unroll
    for (int r = 0; r < 8; ++r) {
      const int qrow_g = qt16 + hi * 8 + r;
      float s0 = S0.f[r] * SCALE;
      float s1 = S1.f[r] * SCALE;
      if (j0 + m      > qrow_g) s0 = -1e30f;   // causal mask
      if (j0 + 16 + m > qrow_g) s1 = -1e30f;
      float rmax = fmaxf(s0, s1);
#pragma unroll
      for (int off = 1; off < 16; off <<= 1)
        rmax = fmaxf(rmax, __shfl_xor(rmax, off, 32));
      float mnew  = fmaxf(m_i[r], rmax);
      float alpha = __expf(m_i[r] - mnew);
      float p0 = __expf(s0 - mnew);
      float p1 = __expf(s1 - mnew);
      float rsum = p0 + p1;
#pragma unroll
      for (int off = 1; off < 16; off <<= 1)
        rsum += __shfl_xor(rsum, off, 32);
      l_i[r] = l_i[r] * alpha + rsum;
      m_i[r] = mnew;
#pragma unroll
      for (int dt = 0; dt < 8; ++dt) O[dt].f[r] *= alpha;
      Plds_s[(hi * 8 + r) * 40 + m]      = f32_to_bf16_raw(p0);
      Plds_s[(hi * 8 + r) * 40 + 16 + m] = f32_to_bf16_raw(p1);
    }

    asm volatile("s_wait_dscnt 0" ::: "memory");   // per-wave DS RAW fence

    AFrag Pf;   // A-layout: lane = query row, element = key (K=32)
    Pf.q[0] = *(const uint4*)&Plds_u[m * 20 +     hi * 4];
    Pf.q[1] = *(const uint4*)&Plds_u[m * 20 + 8 + hi * 4];

#pragma unroll
    for (int dt = 0; dt < 8; ++dt) {
      const unsigned* vr = vT_u + (size_t)(hkv * HD + dt * 16 + m) * 2048
                                + (size_t)b * 1024 + (j0 >> 1);
      AFrag Vf;
      Vf.q[0] = *(const uint4*)(vr +     hi * 4);
      Vf.q[1] = *(const uint4*)(vr + 8 + hi * 4);
      O[dt].v = __builtin_amdgcn_wmma_f32_16x16x32_bf16(false, Pf.v, false, Vf.v,
                                                        (short)0, O[dt].v, false, false);
    }
    asm volatile("s_wait_dscnt 0" ::: "memory");   // WAR before next P scatter
  }

#pragma unroll
  for (int r = 0; r < 8; ++r) {
    const int qrow_g = qt16 + hi * 8 + r;
    const float inv_l = 1.0f / l_i[r];
    size_t base = (size_t)(b * T + qrow_g) * 2048 + h * 128 + m;
#pragma unroll
    for (int dt = 0; dt < 8; ++dt)
      out_bf16[base + dt * 16] = f32_to_bf16_raw(O[dt].f[r] * inv_l);
  }
}

// ---------------------------------------------------------------------------
extern "C" void kernel_launch(void* const* d_in, const int* in_sizes, int n_in,
                              void* d_out, int out_size, void* d_ws, size_t ws_size,
                              hipStream_t stream)
{
  (void)in_sizes; (void)n_in; (void)out_size; (void)ws_size;
  const float* x  = (const float*)d_in[0];
  const float* wq = (const float*)d_in[1];
  const float* wk = (const float*)d_in[2];
  const float* wv = (const float*)d_in[3];
  const float* wo = (const float*)d_in[4];
  float* out = (float*)d_out;

  constexpr int B = 2, T = 2048, H = 16, HKV = 4, HD = 128;
  constexpr int M   = B * T;        // 4096
  constexpr int D   = H * HD;       // 2048
  constexpr int DKV = HKV * HD;     // 512
  constexpr int K   = D;            // contraction dim

  char* ws = (char*)d_ws;
  size_t off = 0;
  unsigned short* qb  = (unsigned short*)(ws + off); off += (size_t)M * D * 2;    // [M,D]
  unsigned short* kb  = (unsigned short*)(ws + off); off += (size_t)M * DKV * 2;  // [M,DKV]
  unsigned short* vtb = (unsigned short*)(ws + off); off += (size_t)M * DKV * 2;  // [DKV,M]
  unsigned short* ab  = (unsigned short*)(ws + off); off += (size_t)M * D * 2;    // [M,D]
  unsigned short* xb  = (unsigned short*)(ws + off); off += (size_t)M * D * 2;    // [M,K]
  unsigned short* wqT = (unsigned short*)(ws + off); off += (size_t)D * K * 2;    // [D,K]
  unsigned short* wkT = (unsigned short*)(ws + off); off += (size_t)DKV * K * 2;  // [DKV,K]
  unsigned short* wvT = (unsigned short*)(ws + off); off += (size_t)DKV * K * 2;  // [DKV,K]
  unsigned short* woT = (unsigned short*)(ws + off); off += (size_t)D * K * 2;    // [D,K]

  dim3 blk(256);
  // 0) one-time bf16 conversions (+weight transposes)
  convert_bf16<<<(M * D / 2 + 255) / 256, blk, 0, stream>>>(x, (unsigned*)xb, M * D / 2);
  convert_transpose_bf16<<<dim3(D   / 32, K / 32), blk, 0, stream>>>(wq, wqT, K, D);
  convert_transpose_bf16<<<dim3(DKV / 32, K / 32), blk, 0, stream>>>(wk, wkT, K, DKV);
  convert_transpose_bf16<<<dim3(DKV / 32, K / 32), blk, 0, stream>>>(wv, wvT, K, DKV);
  convert_transpose_bf16<<<dim3(D   / 32, K / 32), blk, 0, stream>>>(wo, woT, K, D);
  // 1) projections (async-to-LDS double-buffered WMMA GEMMs)
  gemm_wmma_bf16<1, 0><<<dim3(D   / 64, M / 64), blk, 0, stream>>>(xb, wqT, qb,  M, D,   K);
  gemm_wmma_bf16<1, 0><<<dim3(DKV / 64, M / 64), blk, 0, stream>>>(xb, wkT, kb,  M, DKV, K);
  gemm_wmma_bf16<1, 1><<<dim3(DKV / 64, M / 64), blk, 0, stream>>>(xb, wvT, vtb, M, DKV, K);
  // 2) RoPE on Q and K
  rope_kernel<<<(M * H   * 64) / 256, blk, 0, stream>>>(qb, H,   M * H   * 64);
  rope_kernel<<<(M * HKV * 64) / 256, blk, 0, stream>>>(kb, HKV, M * HKV * 64);
  // 3) causal GQA flash attention
  flash_attn_wmma<<<B * H * (T / 16), 32, 0, stream>>>(
      (const unsigned*)qb, (const unsigned*)kb, (const unsigned*)vtb, ab);
  // 4) output projection (f32 out)
  gemm_wmma_bf16<0, 0><<<dim3(D / 64, M / 64), blk, 0, stream>>>(ab, woT, out, M, D, K);
}
